// CrossAttentionFusion_61692910239846
// MI455X (gfx1250) — compile-verified, matrix operand order checked by reading, not run on hardware
//
#include <hip/hip_runtime.h>

// ---------------------------------------------------------------------------
// CrossAttentionFusion collapses: softmax over a length-1 axis == 1, so
//   out = LN( x + swap(x) @ (Wo@Wv)^T + (Wo@bv + bo) ) * gamma + beta
// Memory-bound problem (2.15 GB min traffic, 137 GFLOP) -> f16 WMMA GEMM with
// register-resident folded weights, persistent kernel.
// ---------------------------------------------------------------------------

typedef __attribute__((ext_vector_type(16))) _Float16 v16h;
typedef __attribute__((ext_vector_type(8)))  _Float16 v8h;
typedef __attribute__((ext_vector_type(4)))  _Float16 v4h;
typedef __attribute__((ext_vector_type(8)))  float    v8f;
typedef __attribute__((ext_vector_type(4)))  float    v4f;

#define DD      256          // model dim
#define STRIP   16           // rows per strip (M-tile)
#define NWAVES  16           // one wave per 16-col N-tile
#define BLOCK   (NWAVES * 32)
#define APITCH  (DD + 8)     // f16 LDS pitch (528B: 16B aligned, 4-bank row skew)
#define FPITCH  (DD + 4)     // f32 LDS pitch (1040B: 16B aligned, bank skew)
#define LN_EPS  1e-5f

// ---------------- prep: fold weights once --------------------------------
__global__ void prep_wc_kernel(const float* __restrict__ in_proj_w,
                               const float* __restrict__ out_proj_w,
                               _Float16* __restrict__ wc16) {
    const int n = blockIdx.x;            // output row of Wc
    const int k = threadIdx.x;           // output col of Wc
    const float* Wv = in_proj_w + 2 * DD * DD;   // V-projection block
    float acc = 0.0f;
    for (int j = 0; j < DD; ++j)
        acc = fmaf(out_proj_w[n * DD + j], Wv[j * DD + k], acc);
    wc16[n * DD + k] = (_Float16)acc;
}

__global__ void prep_cb_kernel(const float* __restrict__ in_proj_b,
                               const float* __restrict__ out_proj_b,
                               const float* __restrict__ out_proj_w,
                               float* __restrict__ cb) {
    const int n = threadIdx.x;
    const float* bv = in_proj_b + 2 * DD;
    float acc = out_proj_b[n];
    for (int j = 0; j < DD; ++j)
        acc = fmaf(out_proj_w[n * DD + j], bv[j], acc);
    cb[n] = acc;
}

// ---------------- main fused kernel --------------------------------------
__global__ __launch_bounds__(BLOCK)
void fused_swapmm_ln_kernel(const float* __restrict__ x,
                            const _Float16* __restrict__ wc16,
                            const float* __restrict__ cb,
                            const float* __restrict__ gamma,
                            const float* __restrict__ beta,
                            float* __restrict__ out,
                            int n_rows) {
    __shared__ _Float16 Alds[STRIP][APITCH];   // f16 src rows (swap applied)
    __shared__ float    Xlds[STRIP][FPITCH];   // f32 rows for residual
    __shared__ float    Ylds[STRIP][FPITCH];   // y = x + fused, pre-LN
    __shared__ float    Glds[DD];
    __shared__ float    Blds[DD];

    const int tid   = threadIdx.x;
    const int wave  = tid >> 5;      // 0..15 : N-tile id / LN row id
    const int lane  = tid & 31;
    const int khalf = lane >> 4;     // which K-half of each 16-block (ISA A layout)
    const int nrow  = lane & 15;     // M for A frags, N for B/C frags

    if (tid < DD)            Glds[tid]      = gamma[tid];
    else if (tid < 2 * DD)   Blds[tid - DD] = beta[tid - DD];

    const int   n_off = wave * 16;
    const int   ncol  = n_off + nrow;          // this lane's output column
    const float cbn   = cb[ncol];              // folded bias (column-constant)

    // ---- B fragments: whole 256x16 weight strip, register resident -------
    // B[k][n] = Wc[n][k]; lane layout mirrors the A layout with M -> N.
    v16h bfrag[8];
    {
        const _Float16* wrow = wc16 + (long)ncol * DD;
#pragma unroll
        for (int kk = 0; kk < 8; ++kk) {
            const int k0 = kk * 32 + khalf * 8;
            v8h lo = *(const v8h*)(wrow + k0);
            v8h hi = *(const v8h*)(wrow + k0 + 16);
            bfrag[kk] = __builtin_shufflevector(lo, hi,
                0, 1, 2, 3, 4, 5, 6, 7, 8, 9, 10, 11, 12, 13, 14, 15);
        }
    }

    const int nstrips = n_rows >> 4;
    for (int strip = blockIdx.x; strip < nstrips; strip += gridDim.x) {
        const long base = (long)strip * STRIP;

        // ---- stage strip: x -> Xlds (f32) and Alds (f16, row^1 swap) -----
#pragma unroll
        for (int c = tid; c < STRIP * (DD / 4); c += BLOCK) {
            const int t  = c >> 6;          // strip-local source row
            const int c4 = (c & 63) << 2;   // column
            v4f vx = *(const v4f*)(x + (base + t) * DD + c4);
            *(v4f*)&Xlds[t][c4] = vx;
            v4h vh;
            vh[0] = (_Float16)vx[0]; vh[1] = (_Float16)vx[1];
            vh[2] = (_Float16)vx[2]; vh[3] = (_Float16)vx[3];
            // output row m pulls source row m^1 -> source row t feeds A slot t^1
            *(v4h*)&Alds[t ^ 1][c4] = vh;
        }
        __syncthreads();

        // prefetch next strip of x while the WMMAs run (global_prefetch_b8)
        {
            const long nxt = (long)(strip + gridDim.x);
            if (nxt < nstrips) {
                const char* p = (const char*)(x + nxt * STRIP * DD);
                __builtin_prefetch(p + tid * 32, 0, 0);
            }
        }

        // ---- GEMM: 16x16 tile per wave, K = 256 in 8 WMMA steps ----------
        v8f acc;
#pragma unroll
        for (int i = 0; i < 8; ++i) acc[i] = cbn;

#pragma unroll
        for (int kk = 0; kk < 8; ++kk) {
            const int k0 = kk * 32 + khalf * 8;
            v8h lo = *(const v8h*)&Alds[nrow][k0];
            v8h hi = *(const v8h*)&Alds[nrow][k0 + 16];
            v16h afrag = __builtin_shufflevector(lo, hi,
                0, 1, 2, 3, 4, 5, 6, 7, 8, 9, 10, 11, 12, 13, 14, 15);
            acc = __builtin_amdgcn_wmma_f32_16x16x32_f16(
                false, afrag, false, bfrag[kk], (short)0, acc, false, false);
        }

        // ---- epilogue: y = x + fused (+cb already in acc) -> Ylds --------
        // C layout: lane%16 = N, VGPR v -> M = 8*(lane/16) + v
#pragma unroll
        for (int v = 0; v < 8; ++v) {
            const int m = khalf * 8 + v;
            Ylds[m][ncol] = Xlds[m][ncol] + acc[v];
        }
        __syncthreads();   // also fences next-iter Alds/Xlds writes vs GEMM/epilogue reads

        // ---- LayerNorm: wave w normalizes row w --------------------------
        {
            const int m  = wave;
            const int c0 = lane * 8;
            v4f ya = *(const v4f*)&Ylds[m][c0];
            v4f yb = *(const v4f*)&Ylds[m][c0 + 4];
            float s = 0.0f, s2 = 0.0f;
#pragma unroll
            for (int i = 0; i < 4; ++i) {
                s  += ya[i] + yb[i];
                s2 += ya[i] * ya[i] + yb[i] * yb[i];
            }
#pragma unroll
            for (int off = 16; off > 0; off >>= 1) {
                s  += __shfl_xor(s,  off, 32);
                s2 += __shfl_xor(s2, off, 32);
            }
            const float mu  = s * (1.0f / DD);
            const float var = s2 * (1.0f / DD) - mu * mu;
            const float inv = rsqrtf(var + LN_EPS);

            v4f o0, o1;
#pragma unroll
            for (int i = 0; i < 4; ++i) {
                o0[i] = (ya[i] - mu) * inv * Glds[c0 + i]     + Blds[c0 + i];
                o1[i] = (yb[i] - mu) * inv * Glds[c0 + 4 + i] + Blds[c0 + 4 + i];
            }
            float* orow = out + (base + m) * DD + c0;
            *(v4f*)(orow)     = o0;
            *(v4f*)(orow + 4) = o1;
        }
        // next iteration's __syncthreads() (after staging) orders
        // Ylds overwrite vs this iteration's LN readers.
    }
}

// ---------------------------------------------------------------------------
extern "C" void kernel_launch(void* const* d_in, const int* in_sizes, int n_in,
                              void* d_out, int out_size, void* d_ws, size_t ws_size,
                              hipStream_t stream) {
    const float* x   = (const float*)d_in[0];   // (B, 2, D) f32
    const float* ipw = (const float*)d_in[1];   // (3D, D)
    const float* ipb = (const float*)d_in[2];   // (3D,)
    const float* opw = (const float*)d_in[3];   // (D, D)
    const float* opb = (const float*)d_in[4];   // (D,)
    const float* g   = (const float*)d_in[5];   // (D,)
    const float* b   = (const float*)d_in[6];   // (D,)
    float* out = (float*)d_out;

    _Float16* wc16 = (_Float16*)d_ws;                                 // 128 KB
    float*    cbp  = (float*)((char*)d_ws + DD * DD * sizeof(_Float16)); // 1 KB

    const int n_rows = in_sizes[0] / DD;        // 2*B flattened rows

    prep_wc_kernel<<<DD, DD, 0, stream>>>(ipw, opw, wc16);
    prep_cb_kernel<<<1, DD, 0, stream>>>(ipb, opb, opw, cbp);

    const int nstrips = n_rows / STRIP;
    int grid = nstrips < 2048 ? nstrips : 2048;  // persistent, grid-stride
    if (grid < 1) grid = 1;
    fused_swapmm_ln_kernel<<<grid, BLOCK, 0, stream>>>(x, wc16, cbp, g, b, out, n_rows);
}
